// XLNetLayer_60086592471111
// MI455X (gfx1250) — matrix operand order repaired and checked by reading, not compiled, not run on hardware
//
#include <hip/hip_runtime.h>
#include <math.h>

typedef __bf16 bf16;
typedef __attribute__((ext_vector_type(16))) __bf16 v16bf;
typedef __attribute__((ext_vector_type(8)))  __bf16 v8bf;
typedef __attribute__((ext_vector_type(4)))  __bf16 v4bf;
typedef __attribute__((ext_vector_type(8)))  float  v8f;
typedef __attribute__((ext_vector_type(4)))  float  v4f;

#define QL 1024
#define KL 1024
#define RL 2048
#define NP 256
#define BS 4
#define NH 16
#define DH 64
#define DM 1024
#define DI 4096
#define ATTN_SCALE 0.125f   // 1/sqrt(64)

// ---------------------------------------------------------------------------
// Batched WMMA GEMM with K-contiguous operands (all launches guarantee this):
//   A[m,k] = A[z*sA + m*rsA + k]        (bf16, k contiguous)
//   B[k,n] = B[z*sB + n*csB + k]        (bf16, k contiguous)
//   C[m,n] at z*sC + m*rsC + n*csC      (dtype per EPI)
// EPI: 0 = store f32, 1 = store bf16, 2 = gelu(exact erf) -> bf16,
//      3 = rel-shift scatter-add into f32:  jo = n + m - epiI, 0<=jo<epiK
//
// LDS tiles in FRAGMENT-NATIVE order (ISA per-lane register layout): each
// lane reads each v16bf fragment as two ds_load_b128; each thread stages
// 8-element K-octets with global b128 loads + ds_store_b128. Double-buffered
// LDS (one barrier per K-step); next tile's global loads overlap the WMMAs.
// Block: 256 threads (8 waves). Macro tile 128(M) x 64(N); wave = 32x32
// (2 A-frags x 2 B-frags -> 4 WMMAs per fragment set).
// Requires M%128==0, N%64==0, K%32==0.
// ---------------------------------------------------------------------------
template<int EPI>
__global__ __launch_bounds__(256)
void gemm_wmma(const bf16* __restrict__ A, const bf16* __restrict__ B,
               void* __restrict__ Cv, const float* __restrict__ bias,
               int M, int N, int K,
               long sA, long sB, long sC,
               long rsA, long csB, long rsC, long csC,
               float alpha, int epiI, int epiK)
{
    __shared__ bf16 As[2][8 * 32 * 16];   // [buf][sub16][lane][16]  (128x32)
    __shared__ bf16 Bs[2][4 * 32 * 16];   // [buf][wn*2+sub][lane][16] (32x64)

    const int  t    = threadIdx.x;
    const int  lane = t & 31;
    const int  wave = t >> 5;
    const int  wm   = wave & 3;   // M sub-tile (32 rows)
    const int  wn   = wave >> 2;  // N sub-tile (32 cols)
    const int  lgrp = lane >> 4;  // lane group 0/1
    const int  l16  = lane & 15;
    const int  bz   = blockIdx.z;
    const long m0   = (long)blockIdx.y * 128;
    const long n0   = (long)blockIdx.x * 64;

    // staging assignments (one K-octet per task; A has 2 tasks per thread)
    const int rA = t >> 2, oA = t & 3;     // rows 0..63 (task2: +64)
    const int dstA = ((rA >> 4) * 32 + (oA & 1) * 16 + (rA & 15)) * 16 + (oA >> 1) * 8;
    const int nB = t >> 2, oB = t & 3;
    const int dstB = (((nB >> 5) * 2 + ((nB >> 4) & 1)) * 32 + (oB & 1) * 16 + (nB & 15)) * 16
                     + (oB >> 1) * 8;

    const bf16* Aptr = A + (long)bz * sA + (m0 + rA) * rsA + oA * 8;
    const bf16* Bptr = B + (long)bz * sB + (n0 + nB) * csB + oB * 8;
    const long  a1off = 64 * rsA;          // second A task: rows +64

    v8bf a0 = *(const v8bf*)Aptr;
    v8bf a1 = *(const v8bf*)(Aptr + a1off);
    v8bf b0 = *(const v8bf*)Bptr;
    *(v8bf*)&As[0][dstA]        = a0;
    *(v8bf*)&As[0][dstA + 2048] = a1;      // rows 64..127 -> sub16 4..7
    *(v8bf*)&Bs[0][dstB]        = b0;
    __syncthreads();

    v8f acc00 = {}, acc01 = {}, acc10 = {}, acc11 = {};
    const int steps = K >> 5;

    for (int s = 0; s < steps; ++s) {
        const int cur = s & 1;
        if (s + 1 < steps) {
            a0 = *(const v8bf*)(Aptr + (long)(s + 1) * 32);
            a1 = *(const v8bf*)(Aptr + a1off + (long)(s + 1) * 32);
            b0 = *(const v8bf*)(Bptr + (long)(s + 1) * 32);
        }
        v16bf af0 = *(const v16bf*)&As[cur][((wm * 2 + 0) * 32 + lane) * 16];
        v16bf af1 = *(const v16bf*)&As[cur][((wm * 2 + 1) * 32 + lane) * 16];
        v16bf bf0 = *(const v16bf*)&Bs[cur][((wn * 2 + 0) * 32 + lane) * 16];
        v16bf bf1 = *(const v16bf*)&Bs[cur][((wn * 2 + 1) * 32 + lane) * 16];

        acc00 = __builtin_amdgcn_wmma_f32_16x16x32_bf16(false, af0, false, bf0,
                                                        (short)0, acc00, false, false);
        acc01 = __builtin_amdgcn_wmma_f32_16x16x32_bf16(false, af0, false, bf1,
                                                        (short)0, acc01, false, false);
        acc10 = __builtin_amdgcn_wmma_f32_16x16x32_bf16(false, af1, false, bf0,
                                                        (short)0, acc10, false, false);
        acc11 = __builtin_amdgcn_wmma_f32_16x16x32_bf16(false, af1, false, bf1,
                                                        (short)0, acc11, false, false);
        if (s + 1 < steps) {
            *(v8bf*)&As[cur ^ 1][dstA]        = a0;
            *(v8bf*)&As[cur ^ 1][dstA + 2048] = a1;
            *(v8bf*)&Bs[cur ^ 1][dstB]        = b0;
            __syncthreads();
        }
    }

    // epilogue: C layout — VGPR v: rows v + lgrp*8, col = lane%16
    #pragma unroll
    for (int i = 0; i < 2; ++i) {
        #pragma unroll
        for (int v = 0; v < 8; ++v) {
            long m = m0 + (wm * 2 + i) * 16 + lgrp * 8 + v;
            #pragma unroll
            for (int sub = 0; sub < 2; ++sub) {
                long n = n0 + wn * 32 + sub * 16 + l16;
                float val;
                if (i == 0) val = (sub ? acc01[v] : acc00[v]);
                else        val = (sub ? acc11[v] : acc10[v]);
                val *= alpha;
                if (bias) val += bias[n];
                if (EPI == 0) {
                    ((float*)Cv)[(long)bz * sC + m * rsC + n * csC] = val;
                } else if (EPI == 1) {
                    ((bf16*)Cv)[(long)bz * sC + m * rsC + n * csC] = (bf16)val;
                } else if (EPI == 2) {
                    float ge = 0.5f * val * (1.0f + erff(val * 0.70710678118654752f));
                    ((bf16*)Cv)[(long)bz * sC + m * rsC + n * csC] = (bf16)ge;
                } else {
                    long jo = n + m - (long)epiI;
                    if (jo >= 0 && jo < epiK)
                        ((float*)Cv)[(long)bz * sC + m * rsC + jo * csC] += val;
                }
            }
        }
    }
}

// ---------------------------------------------------------------------------
// f32 -> bf16 conversion, 8 elements/thread (n % 8 == 0)
// ---------------------------------------------------------------------------
__global__ void cvt_f32_bf16(const float* __restrict__ in, bf16* __restrict__ out, long n)
{
    long stride = (long)gridDim.x * 256 * 8;
    for (long i = ((long)blockIdx.x * 256 + threadIdx.x) * 8; i < n; i += stride) {
        v4f a = *(const v4f*)(in + i);
        v4f b = *(const v4f*)(in + i + 4);
        v8bf o;
        #pragma unroll 4
        for (int j = 0; j < 4; ++j) { o[j] = (bf16)a[j]; o[j + 4] = (bf16)b[j]; }
        *(v8bf*)(out + i) = o;
    }
}

// f32 [Kd, Nd] row-major -> bf16 transposed [Nd, Kd] (k-contiguous weights)
__global__ void cvt_transpose(const float* __restrict__ in, bf16* __restrict__ out,
                              int Kd, int Nd)
{
    long total8 = (long)(Kd / 8) * Nd;
    for (long id = (long)blockIdx.x * 256 + threadIdx.x; id < total8;
         id += (long)gridDim.x * 256) {
        int k8 = (int)(id % (Kd / 8));
        int n  = (int)(id / (Kd / 8));
        v8bf o;
        #pragma unroll 8
        for (int j = 0; j < 8; ++j) o[j] = (bf16)in[(long)(k8 * 8 + j) * Nd + n];
        *(v8bf*)&out[(long)n * Kd + k8 * 8] = o;
    }
}

// tm[m,l,b] (f32) -> tmTf[b][l][m] and tmTr[b][m][l] (bf16)
__global__ void transpose_tm(const float* __restrict__ tm,
                             bf16* __restrict__ tmTf, bf16* __restrict__ tmTr)
{
    long total = (long)NP * QL * BS;
    for (long id = (long)blockIdx.x * 256 + threadIdx.x; id < total;
         id += (long)gridDim.x * 256) {
        int  b = (int)(id % BS);
        int  l = (int)((id / BS) % QL);
        int  m = (int)(id / ((long)BS * QL));
        bf16 v = (bf16)tm[id];
        tmTf[((long)b * QL + l) * NP + m] = v;
        tmTr[((long)b * NP + m) * QL + l] = v;
    }
}

// plain [s*BS+b, NH*DH] -> heads [b,nh,s,dh]   (8-wide along dh)
__global__ void permute_heads(const bf16* __restrict__ in, bf16* __restrict__ out, int seq)
{
    long total8 = (long)seq * BS * NH * (DH / 8);
    for (long id = (long)blockIdx.x * 256 + threadIdx.x; id < total8;
         id += (long)gridDim.x * 256) {
        int  d8 = (int)(id % (DH / 8));
        long r  = id / (DH / 8);
        int  s  = (int)(r % seq);
        long r2 = r / seq;
        int  n  = (int)(r2 % NH);
        int  b  = (int)(r2 / NH);
        v8bf val = *(const v8bf*)&in[((long)s * BS + b) * (NH * DH) + n * DH + d8 * 8];
        *(v8bf*)&out[id * 8] = val;
    }
}

// plain [j*BS+b, NH*DH] -> d-major heads [b][nh][dh][j]   (8-wide along j)
__global__ void permute_headsT(const bf16* __restrict__ in, bf16* __restrict__ out, int seq)
{
    long total8 = (long)BS * NH * DH * (seq / 8);
    for (long id = (long)blockIdx.x * 256 + threadIdx.x; id < total8;
         id += (long)gridDim.x * 256) {
        int  j8 = (int)(id % (seq / 8));
        long r  = id / (seq / 8);
        int  d  = (int)(r % DH);
        long r2 = r / DH;
        int  n  = (int)(r2 % NH);
        int  b  = (int)(r2 / NH);
        v8bf o;
        #pragma unroll 8
        for (int jj = 0; jj < 8; ++jj)
            o[jj] = in[((long)(j8 * 8 + jj) * BS + b) * DM + n * DH + d];
        *(v8bf*)&out[id * 8] = o;
    }
}

// plain q -> two biased head-layout copies (q + r_w_bias, q + r_r_bias)
__global__ void qbias_heads(const bf16* __restrict__ in,
                            const float* __restrict__ rwb, const float* __restrict__ rrb,
                            bf16* __restrict__ outRW, bf16* __restrict__ outRR, int seq)
{
    long total8 = (long)seq * BS * NH * (DH / 8);
    for (long id = (long)blockIdx.x * 256 + threadIdx.x; id < total8;
         id += (long)gridDim.x * 256) {
        int  d8 = (int)(id % (DH / 8));
        long r  = id / (DH / 8);
        int  s  = (int)(r % seq);
        long r2 = r / seq;
        int  n  = (int)(r2 % NH);
        int  b  = (int)(r2 / NH);
        int  nd = n * DH + d8 * 8;
        v8bf q = *(const v8bf*)&in[((long)s * BS + b) * (NH * DH) + nd];
        v4f w0 = *(const v4f*)&rwb[nd];
        v4f w1 = *(const v4f*)&rwb[nd + 4];
        v4f r0 = *(const v4f*)&rrb[nd];
        v4f r1 = *(const v4f*)&rrb[nd + 4];
        v8bf ow, orr;
        #pragma unroll 4
        for (int j = 0; j < 4; ++j) {
            float qa = (float)q[j], qb = (float)q[j + 4];
            ow[j]      = (bf16)(qa + w0[j]);
            ow[j + 4]  = (bf16)(qb + w1[j]);
            orr[j]     = (bf16)(qa + r0[j]);
            orr[j + 4] = (bf16)(qb + r1[j]);
        }
        *(v8bf*)&outRW[id * 8] = ow;
        *(v8bf*)&outRR[id * 8] = orr;
    }
}

// masked softmax over a score row [b,nh,i,KL]; writes bf16 probs in place
__global__ __launch_bounds__(256)
void softmax_rows(float* __restrict__ scores, const float* __restrict__ mask)
{
    long row = blockIdx.x;               // (b*NH+n)*QL + i
    int  i   = (int)(row % QL);
    long bn  = row / QL;
    int  b   = (int)(bn / NH);
    float* rp = scores + row * KL;
    const int t  = threadIdx.x;
    const int j0 = t * 4;                // 4 contiguous per thread

    v4f vals = *(const v4f*)(rp + j0);
    float mx = -3.4e38f;
    #pragma unroll 4
    for (int j = 0; j < 4; ++j) {
        vals[j] -= 1e30f * mask[((long)i * KL + j0 + j) * BS + b];
        mx = fmaxf(mx, vals[j]);
    }
    __shared__ float red[256];
    red[t] = mx; __syncthreads();
    for (int off = 128; off; off >>= 1) {
        if (t < off) red[t] = fmaxf(red[t], red[t + off]);
        __syncthreads();
    }
    mx = red[0]; __syncthreads();

    float sum = 0.f;
    #pragma unroll 4
    for (int j = 0; j < 4; ++j) { vals[j] = __expf(vals[j] - mx); sum += vals[j]; }
    red[t] = sum; __syncthreads();
    for (int off = 128; off; off >>= 1) {
        if (t < off) red[t] += red[t + off];
        __syncthreads();
    }
    float inv = 1.f / red[0];
    __syncthreads();

    v4bf p;
    #pragma unroll 4
    for (int j = 0; j < 4; ++j) p[j] = (bf16)(vals[j] * inv);
    *(v4bf*)((bf16*)rp + j0) = p;        // in-place: bf16 row fits in first half
}

// out = LayerNorm(x + res) * gamma + beta ; optional f32 and bf16 outputs
__global__ __launch_bounds__(256)
void ln_residual(const float* __restrict__ x, const float* __restrict__ res,
                 const float* __restrict__ gamma, const float* __restrict__ beta,
                 float* __restrict__ outf, bf16* __restrict__ outb)
{
    long row = blockIdx.x;
    const int t  = threadIdx.x;
    const int j0 = t * 4;

    v4f v  = *(const v4f*)(x   + row * DM + j0);
    v4f rv = *(const v4f*)(res + row * DM + j0);
    float s = 0.f;
    #pragma unroll 4
    for (int j = 0; j < 4; ++j) { v[j] += rv[j]; s += v[j]; }

    __shared__ float red[256];
    red[t] = s; __syncthreads();
    for (int off = 128; off; off >>= 1) { if (t < off) red[t] += red[t + off]; __syncthreads(); }
    float mean = red[0] * (1.f / DM); __syncthreads();

    float s2 = 0.f;
    #pragma unroll 4
    for (int j = 0; j < 4; ++j) { float d = v[j] - mean; s2 += d * d; }
    red[t] = s2; __syncthreads();
    for (int off = 128; off; off >>= 1) { if (t < off) red[t] += red[t + off]; __syncthreads(); }
    float rstd = rsqrtf(red[0] * (1.f / DM) + 1e-8f);

    v4f gm = *(const v4f*)(gamma + j0);
    v4f bt = *(const v4f*)(beta + j0);
    v4f o;
    v4bf ob;
    #pragma unroll 4
    for (int j = 0; j < 4; ++j) {
        o[j]  = (v[j] - mean) * rstd * gm[j] + bt[j];
        ob[j] = (bf16)o[j];
    }
    if (outf) *(v4f*)(outf + row * DM + j0) = o;
    if (outb) *(v4bf*)(outb + row * DM + j0) = ob;
}

// ---------------------------------------------------------------------------
template<int EPI>
static inline void launch_gemm(hipStream_t st, const bf16* A, const bf16* B, void* C,
                               const float* bias, int M, int N, int K, int batch,
                               long sA, long sB, long sC,
                               long rsA, long csB, long rsC, long csC,
                               float alpha, int epiI = 0, int epiK = 0)
{
    dim3 g(N / 64, M / 128, batch), b(256);
    gemm_wmma<EPI><<<g, b, 0, st>>>(A, B, C, bias, M, N, K, sA, sB, sC,
                                    rsA, csB, rsC, csC, alpha, epiI, epiK);
}

extern "C" void kernel_launch(void* const* d_in, const int* in_sizes, int n_in,
                              void* d_out, int out_size, void* d_ws, size_t ws_size,
                              hipStream_t stream)
{
    (void)in_sizes; (void)n_in; (void)out_size; (void)ws_size;
    const float* h    = (const float*)d_in[0];
    const float* g    = (const float*)d_in[1];
    const float* mh   = (const float*)d_in[2];
    const float* mg   = (const float*)d_in[3];
    const float* r    = (const float*)d_in[4];
    const float* tm   = (const float*)d_in[5];
    const float* qw   = (const float*)d_in[6];
    const float* kw   = (const float*)d_in[7];
    const float* vw   = (const float*)d_in[8];
    const float* rw   = (const float*)d_in[9];
    const float* rrb  = (const float*)d_in[10];
    const float* rwb  = (const float*)d_in[11];
    const float* alns = (const float*)d_in[12];
    const float* alnb = (const float*)d_in[13];
    const float* w1   = (const float*)d_in[14];
    const float* b1   = (const float*)d_in[15];
    const float* w2   = (const float*)d_in[16];
    const float* b2   = (const float*)d_in[17];
    const float* flns = (const float*)d_in[18];
    const float* flnb = (const float*)d_in[19];

    // ---- workspace carve-out -------------------------------------------
    char* wp = (char*)d_ws;
    auto alloc = [&](size_t bytes) -> void* {
        void* ret = (void*)wp;
        wp += (bytes + 255) & ~(size_t)255;
        return ret;
    };
    const long HB = (long)QL * BS;     // 4096 rows of h
    const long GB = (long)NP * BS;     // 1024 rows of g
    const long RB = (long)RL * BS;     // 8192 rows of r

    bf16* hb     = (bf16*)alloc(HB * DM * 2);
    bf16* gb     = (bf16*)alloc(GB * DM * 2);
    bf16* rb     = (bf16*)alloc(RB * DM * 2);
    bf16* tmTf   = (bf16*)alloc((long)BS * QL * NP * 2);   // [b][l][m]
    bf16* tmTr   = (bf16*)alloc((long)BS * NP * QL * 2);   // [b][m][l]
    bf16* wqT    = (bf16*)alloc((long)DM * DM * 2);        // [n][k]
    bf16* wkT    = (bf16*)alloc((long)DM * DM * 2);
    bf16* wvT    = (bf16*)alloc((long)DM * DM * 2);
    bf16* wrT    = (bf16*)alloc((long)DM * DM * 2);
    bf16* w1T    = (bf16*)alloc((long)DI * DM * 2);        // [DI][DM]
    bf16* w2T    = (bf16*)alloc((long)DM * DI * 2);        // [DM][DI]
    bf16* qpl    = (bf16*)alloc(HB * DM * 2);
    bf16* kpl    = (bf16*)alloc(HB * DM * 2);
    bf16* vpl    = (bf16*)alloc(HB * DM * 2);
    bf16* rpl    = (bf16*)alloc(RB * DM * 2);
    bf16* qgplT  = (bf16*)alloc((long)BS * DM * NP * 2);   // [b][c][m]
    bf16* qgmap  = (bf16*)alloc(HB * DM * 2);              // plain [l*BS+b, DM]
    bf16* kh     = (bf16*)alloc((long)BS * NH * KL * DH * 2);  // [b][n][j][d]
    bf16* vhT    = (bf16*)alloc((long)BS * NH * DH * KL * 2);  // [b][n][d][j]
    bf16* kr     = (bf16*)alloc((long)BS * NH * RL * DH * 2);  // [b][n][jr][d]
    bf16* qrwh   = (bf16*)alloc((long)BS * NH * QL * DH * 2);
    bf16* qrrh   = (bf16*)alloc((long)BS * NH * QL * DH * 2);
    bf16* qrwg   = (bf16*)alloc((long)BS * NH * QL * DH * 2);
    bf16* qrrg   = (bf16*)alloc((long)BS * NH * QL * DH * 2);
    bf16* vglT   = (bf16*)alloc((long)BS * DM * QL * 2);   // [b][c][l]
    bf16* outhb  = (bf16*)alloc(HB * DM * 2);
    bf16* outgb  = (bf16*)alloc(GB * DM * 2);
    float* vechf  = (float*)alloc(HB * DM * 4);
    float* vecgmf = (float*)alloc(GB * DM * 4);
    float* outhf  = (float*)alloc(HB * DM * 4);
    float* outgf  = (float*)alloc(GB * DM * 4);
    float* scores = (float*)alloc((long)BS * NH * QL * KL * 4);   // 256 MB, reused
    // FFN buffers alias the scores region (scores are dead by then)
    bf16*  hidh = (bf16*)((char*)scores);
    bf16*  hidg = (bf16*)((char*)scores + (long)HB * DI * 2);
    float* ffoh = (float*)((char*)scores + (long)HB * DI * 2 + (long)GB * DI * 2);
    float* ffog = (float*)((char*)ffoh + (long)HB * DM * 4);

    dim3 B256(256);
    auto cvt = [&](const float* src, bf16* dst, long n) {
        cvt_f32_bf16<<<dim3(1024), B256, 0, stream>>>(src, dst, n);
    };
    auto cvtT = [&](const float* src, bf16* dst, int Kd, int Nd) {
        cvt_transpose<<<dim3(1024), B256, 0, stream>>>(src, dst, Kd, Nd);
    };

    // ---- 1) fp32 -> bf16 operands (weights transposed to k-contiguous) --
    cvt(h, hb, HB * DM);
    cvt(g, gb, GB * DM);
    cvt(r, rb, RB * DM);
    transpose_tm<<<dim3(1024), B256, 0, stream>>>(tm, tmTf, tmTr);
    cvtT(qw, wqT, DM, DM);
    cvtT(kw, wkT, DM, DM);
    cvtT(vw, wvT, DM, DM);
    cvtT(rw, wrT, DM, DM);
    cvtT(w1, w1T, DM, DI);
    cvtT(w2, w2T, DI, DM);

    // ---- 2) projections --------------------------------------------------
    launch_gemm<1>(stream, hb, wkT, kpl, nullptr, (int)HB, DM, DM, 1,
                   0, 0, 0, DM, DM, DM, 1, 1.f);
    launch_gemm<1>(stream, hb, wvT, vpl, nullptr, (int)HB, DM, DM, 1,
                   0, 0, 0, DM, DM, DM, 1, 1.f);
    launch_gemm<1>(stream, hb, wqT, qpl, nullptr, (int)HB, DM, DM, 1,
                   0, 0, 0, DM, DM, DM, 1, 1.f);
    launch_gemm<1>(stream, rb, wrT, rpl, nullptr, (int)RB, DM, DM, 1,
                   0, 0, 0, DM, DM, DM, 1, 1.f);
    // g projection as batch-BS, epilogue writes transposed [b][c][m]
    launch_gemm<1>(stream, gb, wqT, qgplT, nullptr, NP, DM, DM, BS,
                   /*sA*/DM, /*sB*/0, /*sC*/(long)DM * NP,
                   /*rsA*/(long)BS * DM, /*csB*/DM, /*rsC*/1, /*csC*/NP, 1.f);

    // ---- 3) head layouts + biases ---------------------------------------
    permute_heads <<<dim3(2048), B256, 0, stream>>>(kpl, kh,  KL);
    permute_headsT<<<dim3(2048), B256, 0, stream>>>(vpl, vhT, KL);
    permute_heads <<<dim3(2048), B256, 0, stream>>>(rpl, kr,  RL);
    qbias_heads   <<<dim3(2048), B256, 0, stream>>>(qpl, rwb, rrb, qrwh, qrrh, QL);

    // ---- 4) forward mapping: qg_map[l,b,:] = sum_m tm[m,l,b]*qg[m,b,:]
    launch_gemm<1>(stream, tmTf, qgplT, qgmap, nullptr, QL, DM, NP, BS,
                   (long)QL * NP, (long)DM * NP, (long)DM,
                   NP, NP, (long)BS * DM, 1, 1.f);
    qbias_heads<<<dim3(2048), B256, 0, stream>>>(qgmap, rwb, rrb, qrwg, qrrg, QL);

    const long sQ  = (long)QL * DH;
    const long sKH = (long)KL * DH;
    const long sKR = (long)RL * DH;
    const long sSC = (long)QL * KL;

    // ================= h-stream attention ===============================
    launch_gemm<0>(stream, qrwh, kh, scores, nullptr, QL, KL, DH, BS * NH,
                   sQ, sKH, sSC, DH, DH, KL, 1, ATTN_SCALE);
    launch_gemm<3>(stream, qrrh, kr, scores, nullptr, QL, RL, DH, BS * NH,
                   sQ, sKR, sSC, DH, DH, KL, 1, ATTN_SCALE, QL, KL);
    softmax_rows<<<dim3(BS * NH * QL), B256, 0, stream>>>(scores, mh);
    // probs bf16 in-place (row stride 2*KL elems); B = d-major V heads
    launch_gemm<0>(stream, (const bf16*)scores, vhT, vechf, nullptr, QL, DH, KL, BS * NH,
                   2 * sSC, (long)DH * KL, (long)DH, 2L * KL, KL, (long)BS * DM, 1, 1.f);
    ln_residual<<<dim3((int)HB), B256, 0, stream>>>(vechf, h, alns, alnb, outhf, outhb);

    // ================= g-stream attention ===============================
    launch_gemm<0>(stream, qrwg, kh, scores, nullptr, QL, KL, DH, BS * NH,
                   sQ, sKH, sSC, DH, DH, KL, 1, ATTN_SCALE);
    launch_gemm<3>(stream, qrrg, kr, scores, nullptr, QL, RL, DH, BS * NH,
                   sQ, sKR, sSC, DH, DH, KL, 1, ATTN_SCALE, QL, KL);
    softmax_rows<<<dim3(BS * NH * QL), B256, 0, stream>>>(scores, mg);
    // attn_vec_g written l-contiguous: [b][c][l]  (base b*DM+n*DH = 64*z)
    launch_gemm<1>(stream, (const bf16*)scores, vhT, vglT, nullptr, QL, DH, KL, BS * NH,
                   2 * sSC, (long)DH * KL, (long)DH * QL, 2L * KL, KL, 1, QL, 1.f);
    // reverse mapping: out[m,b,:] = sum_l tm[m,l,b] * vec[l,b,:]
    launch_gemm<0>(stream, tmTr, vglT, vecgmf, nullptr, NP, DM, QL, BS,
                   (long)NP * QL, (long)DM * QL, (long)DM,
                   QL, QL, (long)BS * DM, 1, 1.f);
    ln_residual<<<dim3((int)GB), B256, 0, stream>>>(vecgmf, g, alns, alnb, outgf, outgb);

    // ================= feed-forward =====================================
    float* out_h = (float*)d_out;
    float* out_g = out_h + HB * DM;

    launch_gemm<2>(stream, outhb, w1T, hidh, b1, (int)HB, DI, DM, 1,
                   0, 0, 0, DM, DM, DI, 1, 1.f);
    launch_gemm<0>(stream, hidh, w2T, ffoh, b2, (int)HB, DM, DI, 1,
                   0, 0, 0, DI, DI, DM, 1, 1.f);
    ln_residual<<<dim3((int)HB), B256, 0, stream>>>(ffoh, outhf, flns, flnb, out_h, nullptr);

    launch_gemm<2>(stream, outgb, w1T, hidg, b1, (int)GB, DI, DM, 1,
                   0, 0, 0, DM, DM, DI, 1, 1.f);
    launch_gemm<0>(stream, hidg, w2T, ffog, b2, (int)GB, DM, DI, 1,
                   0, 0, 0, DI, DI, DM, 1, 1.f);
    ln_residual<<<dim3((int)GB), B256, 0, stream>>>(ffog, outgf, flns, flnb, out_g, nullptr);
}
